// MultiHeadedAttention_69234872812308
// MI455X (gfx1250) — matrix operand order repaired and checked by reading, not compiled, not run on hardware
//
#include <hip/hip_runtime.h>
#include <hip/hip_bf16.h>

// ---------------------------------------------------------------------------
// MI455X (gfx1250) multi-head attention, bf16 WMMA with fp32 accumulation.
// B=4, S=2048, D=1024, H=16, DK=64.
// Round 3: GEMM k-loop ping-pong unrolled 2x (no register-copy rotation, the
// v_dual_mov_b32 storm from round 2 is gone); prefetch locality fixed so
// global_prefetch_b8 targets near caches instead of SCOPE_SYS.
// ---------------------------------------------------------------------------

typedef __bf16 bf16;
typedef __attribute__((ext_vector_type(16))) __bf16 v16bf;
typedef __attribute__((ext_vector_type(8)))  float  v8f;

constexpr int CB  = 4;
constexpr int CS  = 2048;
constexpr int CD  = 1024;
constexpr int CH  = 16;
constexpr int CDK = 64;
// (1/sqrt(DK)) * log2(e): scores kept in log2 domain so softmax uses exp2f.
constexpr float kScale = 0.125f * 1.4426950408889634f;

union Frag2 { uint4 u[2]; v16bf v; };
union Pack8 { uint4 u; bf16 h[8]; };

__device__ inline v16bf load_frag(const bf16* lo, const bf16* hi) {
  Frag2 f;
  f.u[0] = *reinterpret_cast<const uint4*>(lo);
  f.u[1] = *reinterpret_cast<const uint4*>(hi);
  return f.v;
}

__device__ inline v8f wmma_bf16(v16bf a, v16bf b, v8f c) {
  // D = A(16x32) * B(32x16) + C, fp32 accumulate.
  return __builtin_amdgcn_wmma_f32_16x16x32_bf16(
      /*neg_a=*/false, a, /*neg_b=*/false, b,
      /*c_mod=*/(short)0, c, /*reuse_a=*/false, /*reuse_b=*/false);
}

__device__ inline v8f v8f_zero() {
  v8f z = {0.f, 0.f, 0.f, 0.f, 0.f, 0.f, 0.f, 0.f};
  return z;
}

// K fragments for one 32-key chunk: [tile0 dk0-31, tile0 dk32-63,
//                                    tile1 dk0-31, tile1 dk32-63]
__device__ inline void load_k_frags(const bf16* kb0, v16bf kf[4]) {
  const bf16* kb1 = kb0 + 16 * CDK;
  kf[0] = load_frag(kb0, kb0 + 8);
  kf[1] = load_frag(kb0 + 32, kb0 + 40);
  kf[2] = load_frag(kb1, kb1 + 8);
  kf[3] = load_frag(kb1 + 32, kb1 + 40);
}

// V fragments (transposed V: contraction dim contiguous), 4 dk-tiles.
__device__ inline void load_v_frags(const bf16* vb, v16bf vf[4]) {
#pragma unroll
  for (int t = 0; t < 4; ++t) {
    const bf16* vtp = vb + (size_t)t * 16 * CS;
    vf[t] = load_frag(vtp, vtp + 8);
  }
}

// ---------------------------------------------------------------------------
// Stage 0: fp32 -> bf16 conversion (x and weight matrices).
// ---------------------------------------------------------------------------
__global__ __launch_bounds__(256) void cvt_f32_bf16(
    const float* __restrict__ in, bf16* __restrict__ out, int n) {
  int i = blockIdx.x * blockDim.x + threadIdx.x;
  const int stride = gridDim.x * blockDim.x;
  for (; i < n; i += stride) out[i] = (bf16)in[i];
}

// ---------------------------------------------------------------------------
// Stage 1: fused QKV projection.  Y = x @ W^T + b, x:(B*S, D) bf16.
// One wave computes a 16x64 output strip; ping-pong double-buffered
// fragments (2x k-unroll, copy-free).  blockIdx.z selects {Wq,Wk,Wv}.
// Q,K written (B,H,S,DK); V written transposed (B,H,DK,S).
// ---------------------------------------------------------------------------
__global__ __launch_bounds__(256) void qkv_proj(
    const bf16* __restrict__ xh,
    const bf16* __restrict__ Wqh, const bf16* __restrict__ Wkh,
    const bf16* __restrict__ Wvh,
    const float* __restrict__ bq, const float* __restrict__ bk,
    const float* __restrict__ bv,
    bf16* __restrict__ Q, bf16* __restrict__ K, bf16* __restrict__ Vt) {
  const int wave = threadIdx.x >> 5;
  const int lane = threadIdx.x & 31;
  const int ll = lane & 15, hl = lane >> 4;
  const int task = blockIdx.x * 8 + wave;   // 8192 tasks
  const int mtile = task >> 4;              // 0..511 (rows of B*S)
  const int nstrip = task & 15;             // 0..15  (64-col strips)
  const int z = blockIdx.z;

  const bf16* W = (z == 0) ? Wqh : (z == 1) ? Wkh : Wvh;
  const float* bias = (z == 0) ? bq : (z == 1) ? bk : bv;

  v8f acc[4];
#pragma unroll
  for (int t = 0; t < 4; ++t) acc[t] = v8f_zero();

  const bf16* arow = xh + (size_t)(mtile * 16 + ll) * CD;
  const bf16* brow0 = W + (size_t)(nstrip * 64 + ll) * CD;

  // ---- ping-pong pipelined main loop (2 k-steps per iteration) ----
  v16bf aA = load_frag(arow + hl * 16, arow + hl * 16 + 8);
  v16bf bA[4];
#pragma unroll
  for (int t = 0; t < 4; ++t) {
    const bf16* br = brow0 + (size_t)t * 16 * CD + hl * 16;
    bA[t] = load_frag(br, br + 8);
  }

  for (int k0 = 0; k0 < CD; k0 += 64) {
    const int k1 = k0 + 32;
    v16bf aB = load_frag(arow + k1 + hl * 16, arow + k1 + hl * 16 + 8);
    v16bf bB[4];
#pragma unroll
    for (int t = 0; t < 4; ++t) {
      const bf16* br = brow0 + (size_t)t * 16 * CD + k1 + hl * 16;
      bB[t] = load_frag(br, br + 8);
    }
    __builtin_prefetch(arow + k1 + 32, 0, 3);   // global_prefetch_b8 (near)
#pragma unroll
    for (int t = 0; t < 4; ++t) acc[t] = wmma_bf16(aA, bA[t], acc[t]);

    const int k2 = (k0 + 64 < CD) ? (k0 + 64) : k0;  // dummy reload last iter
    aA = load_frag(arow + k2 + hl * 16, arow + k2 + hl * 16 + 8);
#pragma unroll
    for (int t = 0; t < 4; ++t) {
      const bf16* br = brow0 + (size_t)t * 16 * CD + k2 + hl * 16;
      bA[t] = load_frag(br, br + 8);
    }
#pragma unroll
    for (int t = 0; t < 4; ++t) acc[t] = wmma_bf16(aB, bB[t], acc[t]);
  }

  const int r0 = mtile * 16 + 8 * hl;   // output row for element e: r0+e
  const int bb = r0 / CS, s0 = r0 % CS;
#pragma unroll
  for (int t = 0; t < 4; ++t) {
    const int c = nstrip * 64 + t * 16 + ll;
    const int hh = c / CDK, dd = c % CDK;
    const float bs = bias[c];
    if (z != 2) {
      bf16* out = (z == 0) ? Q : K;
      bf16* o = out + (((size_t)bb * CH + hh) * CS + s0) * CDK + dd;
#pragma unroll
      for (int e = 0; e < 8; ++e) o[(size_t)e * CDK] = (bf16)(acc[t][e] + bs);
    } else {
      // transposed V: 8 consecutive s-values per lane -> one b128 store
      Pack8 p;
#pragma unroll
      for (int e = 0; e < 8; ++e) p.h[e] = (bf16)(acc[t][e] + bs);
      *reinterpret_cast<uint4*>(
          Vt + (((size_t)bb * CH + hh) * CDK + dd) * CS + s0) = p.u;
    }
  }
}

// ---------------------------------------------------------------------------
// Stage 2: flash attention.  One wave owns a 16-query tile of one (b,h).
// Per 32-key chunk: 4 score WMMAs; then V (this chunk) + K (next chunk)
// fragment loads are issued BEFORE the softmax VALU section so VMEM overlaps
// the exp2/shfl work; P routed D-layout -> A-layout through a wave-private
// LDS slab (ds stores + ds_load_b128, s_wait_dscnt); 4 PV WMMAs.
// ---------------------------------------------------------------------------
__global__ __launch_bounds__(256) void attn(
    const bf16* __restrict__ Q, const bf16* __restrict__ K,
    const bf16* __restrict__ Vt, const int* __restrict__ mask,
    bf16* __restrict__ O) {
  __shared__ bf16 plds[8][16 * 32];   // 1KB per wave
  const int wave = threadIdx.x >> 5;
  const int lane = threadIdx.x & 31;
  const int ll = lane & 15, hl = lane >> 4;
  const int task = blockIdx.x * 8 + wave;   // 8192 tasks
  const int bh = task >> 7;                 // 0..63
  const int qt = task & 127;                // 0..127
  const int b = bh >> 4, h = bh & 15;

  const bf16* qbase = Q + ((size_t)bh * CS + qt * 16 + ll) * CDK;
  const v16bf aQ0 = load_frag(qbase + hl * 16, qbase + hl * 16 + 8);
  const v16bf aQ1 = load_frag(qbase + 32 + hl * 16, qbase + 40 + hl * 16);

  v8f accO[4];
#pragma unroll
  for (int t = 0; t < 4; ++t) accO[t] = v8f_zero();
  float mrow[8], lsum[8];
#pragma unroll
  for (int e = 0; e < 8; ++e) { mrow[e] = -1.0e30f; lsum[e] = 0.f; }

  const int* mbase = mask + ((size_t)b * CS + qt * 16 + 8 * hl) * CS;
  const bf16* Kbase = K + ((size_t)bh * CS + ll) * CDK + hl * 16;
  const bf16* Vbase = Vt + ((size_t)bh * CDK + ll) * CS + hl * 16;

  v16bf kf[4], vf[4];
  load_k_frags(Kbase, kf);   // chunk 0

  for (int kv0 = 0; kv0 < CS; kv0 += 32) {
    // ---- scores: Q(16x64) @ K^T for two 16-key tiles ----
    v8f s0 = v8f_zero(), s1 = v8f_zero();
    s0 = wmma_bf16(aQ0, kf[0], s0);
    s0 = wmma_bf16(aQ1, kf[1], s0);
    s1 = wmma_bf16(aQ0, kf[2], s1);
    s1 = wmma_bf16(aQ1, kf[3], s1);

    // ---- issue V (this chunk) and K (next chunk) loads early: they fly
    //      while the softmax VALU below executes ----
    load_v_frags(Vbase + kv0, vf);
    const int kvn = (kv0 + 32 < CS) ? (kv0 + 32) : kv0;
    load_k_frags(Kbase + (size_t)kvn * CDK, kf);

    // ---- scale (log2 domain) + mask ----
    float vv0[8], vv1[8];
    const int* mp = mbase + kv0 + ll;
#pragma unroll
    for (int e = 0; e < 8; ++e) {
      float a0 = s0[e] * kScale;
      float a1 = s1[e] * kScale;
      if (mp[(size_t)e * CS] == 0) a0 = -1.0e9f;
      if (mp[(size_t)e * CS + 16] == 0) a1 = -1.0e9f;
      vv0[e] = a0; vv1[e] = a1;
    }
    // ---- online softmax update ----
#pragma unroll
    for (int e = 0; e < 8; ++e) {
      float rm = fmaxf(vv0[e], vv1[e]);
#pragma unroll
      for (int off = 1; off < 16; off <<= 1)
        rm = fmaxf(rm, __shfl_xor(rm, off, 32));
      const float mn = fmaxf(mrow[e], rm);
      const float corr = exp2f(mrow[e] - mn);
      mrow[e] = mn;
      const float p0 = exp2f(vv0[e] - mn);
      const float p1 = exp2f(vv1[e] - mn);
      lsum[e] = lsum[e] * corr + p0 + p1;
#pragma unroll
      for (int t = 0; t < 4; ++t) accO[t][e] *= corr;
      // D-layout scatter of P into the wave-private LDS slab
      plds[wave][(8 * hl + e) * 32 + ll] = (bf16)p0;
      plds[wave][(8 * hl + e) * 32 + 16 + ll] = (bf16)p1;
    }
    asm volatile("s_wait_dscnt 0" ::: "memory");
    // A-layout gather of P (row = ll, contiguous k per lane)
    const bf16* pb = &plds[wave][(size_t)ll * 32 + hl * 16];
    const v16bf aP = load_frag(pb, pb + 8);
    // ---- accO += P(16x32) @ V(32x64) ----
#pragma unroll
    for (int t = 0; t < 4; ++t) accO[t] = wmma_bf16(aP, vf[t], accO[t]);
  }

  // ---- normalize and store (B,S,H*DK) bf16 for the output projection ----
#pragma unroll
  for (int e = 0; e < 8; ++e) {
    float l = lsum[e];
#pragma unroll
    for (int off = 1; off < 16; off <<= 1) l += __shfl_xor(l, off, 32);
    const float inv = 1.0f / fmaxf(l, 1e-30f);
#pragma unroll
    for (int t = 0; t < 4; ++t) accO[t][e] *= inv;
  }
  const int qrow0 = qt * 16 + 8 * hl;
#pragma unroll
  for (int t = 0; t < 4; ++t) {
    bf16* ob = O + ((size_t)b * CS + qrow0) * CD + h * CDK + t * 16 + ll;
#pragma unroll
    for (int e = 0; e < 8; ++e) ob[(size_t)e * CD] = (bf16)accO[t][e];
  }
}

// ---------------------------------------------------------------------------
// Stage 3: output projection.  out = O @ Wo^T + bo, fp32 result.
// Same ping-pong pipeline as stage 1.
// ---------------------------------------------------------------------------
__global__ __launch_bounds__(256) void out_proj(
    const bf16* __restrict__ Ah, const bf16* __restrict__ Wh,
    const float* __restrict__ bias, float* __restrict__ out) {
  const int wave = threadIdx.x >> 5;
  const int lane = threadIdx.x & 31;
  const int ll = lane & 15, hl = lane >> 4;
  const int task = blockIdx.x * 8 + wave;
  const int mtile = task >> 4;
  const int nstrip = task & 15;

  v8f acc[4];
#pragma unroll
  for (int t = 0; t < 4; ++t) acc[t] = v8f_zero();

  const bf16* arow = Ah + (size_t)(mtile * 16 + ll) * CD;
  const bf16* brow0 = Wh + (size_t)(nstrip * 64 + ll) * CD;

  v16bf aA = load_frag(arow + hl * 16, arow + hl * 16 + 8);
  v16bf bA[4];
#pragma unroll
  for (int t = 0; t < 4; ++t) {
    const bf16* br = brow0 + (size_t)t * 16 * CD + hl * 16;
    bA[t] = load_frag(br, br + 8);
  }

  for (int k0 = 0; k0 < CD; k0 += 64) {
    const int k1 = k0 + 32;
    v16bf aB = load_frag(arow + k1 + hl * 16, arow + k1 + hl * 16 + 8);
    v16bf bB[4];
#pragma unroll
    for (int t = 0; t < 4; ++t) {
      const bf16* br = brow0 + (size_t)t * 16 * CD + k1 + hl * 16;
      bB[t] = load_frag(br, br + 8);
    }
    __builtin_prefetch(arow + k1 + 32, 0, 3);
#pragma unroll
    for (int t = 0; t < 4; ++t) acc[t] = wmma_bf16(aA, bA[t], acc[t]);

    const int k2 = (k0 + 64 < CD) ? (k0 + 64) : k0;
    aA = load_frag(arow + k2 + hl * 16, arow + k2 + hl * 16 + 8);
#pragma unroll
    for (int t = 0; t < 4; ++t) {
      const bf16* br = brow0 + (size_t)t * 16 * CD + k2 + hl * 16;
      bA[t] = load_frag(br, br + 8);
    }
#pragma unroll
    for (int t = 0; t < 4; ++t) acc[t] = wmma_bf16(aB, bB[t], acc[t]);
  }

  const int r0 = mtile * 16 + 8 * hl;
#pragma unroll
  for (int t = 0; t < 4; ++t) {
    const int c = nstrip * 64 + t * 16 + ll;
    const float bs = bias[c];
    float* o = out + (size_t)r0 * CD + c;
#pragma unroll
    for (int e = 0; e < 8; ++e) o[(size_t)e * CD] = acc[t][e] + bs;
  }
}

// ---------------------------------------------------------------------------
// Host launcher.
// Inputs: x, mask, Wq, bq, Wk, bk, Wv, bv, Wo, bo.
// Workspace: xh | Wqh | Wkh | Wvh | Woh | Q | K | Vt | O  (all bf16, ~92 MB)
// ---------------------------------------------------------------------------
extern "C" void kernel_launch(void* const* d_in, const int* in_sizes, int n_in,
                              void* d_out, int out_size, void* d_ws,
                              size_t ws_size, hipStream_t stream) {
  (void)in_sizes; (void)n_in; (void)out_size; (void)ws_size;
  const float* x  = (const float*)d_in[0];
  const int* mask = (const int*)d_in[1];
  const float* Wq = (const float*)d_in[2];
  const float* bq = (const float*)d_in[3];
  const float* Wk = (const float*)d_in[4];
  const float* bk = (const float*)d_in[5];
  const float* Wv = (const float*)d_in[6];
  const float* bv = (const float*)d_in[7];
  const float* Wo = (const float*)d_in[8];
  const float* bo = (const float*)d_in[9];
  float* out = (float*)d_out;

  const size_t nX = (size_t)CB * CS * CD;   // 8,388,608
  const size_t nW = (size_t)CD * CD;        // 1,048,576
  bf16* ws  = (bf16*)d_ws;
  bf16* xh  = ws;  ws += nX;
  bf16* Wqh = ws;  ws += nW;
  bf16* Wkh = ws;  ws += nW;
  bf16* Wvh = ws;  ws += nW;
  bf16* Woh = ws;  ws += nW;
  bf16* Qb  = ws;  ws += nX;
  bf16* Kb  = ws;  ws += nX;
  bf16* Vtb = ws;  ws += nX;
  bf16* Ob  = ws;  ws += nX;

  cvt_f32_bf16<<<512, 256, 0, stream>>>(x,  xh,  (int)nX);
  cvt_f32_bf16<<<128, 256, 0, stream>>>(Wq, Wqh, (int)nW);
  cvt_f32_bf16<<<128, 256, 0, stream>>>(Wk, Wkh, (int)nW);
  cvt_f32_bf16<<<128, 256, 0, stream>>>(Wv, Wvh, (int)nW);
  cvt_f32_bf16<<<128, 256, 0, stream>>>(Wo, Woh, (int)nW);

  qkv_proj<<<dim3(1024, 1, 3), 256, 0, stream>>>(
      xh, Wqh, Wkh, Wvh, bq, bk, bv, Qb, Kb, Vtb);
  attn<<<dim3(1024), 256, 0, stream>>>(Qb, Kb, Vtb, mask, Ob);
  out_proj<<<dim3(1024), 256, 0, stream>>>(Ob, Woh, bo, out);
}
